// MambaBlock_78932908966460
// MI455X (gfx1250) — compile-verified, matrix operand order checked by reading, not run on hardware
//
#include <hip/hip_runtime.h>
#include <hip/hip_bf16.h>

// ---------------------------------------------------------------------------
// Mamba block for MI455X (gfx1250, wave32, WMMA).
// GEMMs via v_wmma_f32_16x16x32_f16 (f16 in, f32 accumulate), 32x32 tile/wave.
// Selective scan: 16 SSM states -> 16 lanes, fused C-contraction + gating.
// ---------------------------------------------------------------------------

typedef _Float16 half8  __attribute__((ext_vector_type(8)));
typedef _Float16 half16 __attribute__((ext_vector_type(16)));
typedef float    float8 __attribute__((ext_vector_type(8)));

#define B_DIM   2
#define L_DIM   2048
#define DMODEL  512
#define DINNER  1024
#define NSTATE  16
#define DTRANK  32
#define MROWS   (B_DIM * L_DIM)   // 4096

// ---------------------------------------------------------------------------
// WMMA 16x16x32 f16 fragment loader (ISA 7.12.2):
//  lane = hi*16 + low ; row/col = base + low
//  elements 0..7  hold K = k0 + hi*8 + i
//  elements 8..15 hold K = k0 + 16 + hi*8 + i
// Two contiguous 16B loads per fragment.
// ---------------------------------------------------------------------------
__device__ __forceinline__ half16 load_frag_h(const _Float16* __restrict__ base,
                                              int ld, int rc, int k0, int hi) {
  const _Float16* p = base + (size_t)rc * ld + k0 + hi * 8;
  half8 lo = *(const half8*)p;
  half8 hh = *(const half8*)(p + 16);
  half16 v;
#pragma unroll
  for (int i = 0; i < 8; ++i) { v[i] = lo[i]; v[i + 8] = hh[i]; }
  return v;
}

// branch-free numerically-stable softplus using HW transcendentals
__device__ __forceinline__ float softplus_fast(float x) {
  return fmaxf(x, 0.0f) + __logf(1.0f + __expf(-fabsf(x)));
}

__device__ __forceinline__ float silu_fast(float x) {
  return x / (1.0f + __expf(-x));
}

// ---------------------------------------------------------------------------
// Elementwise helpers
// ---------------------------------------------------------------------------
__global__ void k_f32_to_f16(const float* __restrict__ src,
                             _Float16* __restrict__ dst, int n) {
  for (int i = blockIdx.x * blockDim.x + threadIdx.x; i < n;
       i += gridDim.x * blockDim.x)
    dst[i] = (_Float16)src[i];
}

__global__ void k_a_neg(const float* __restrict__ a_log,
                        float* __restrict__ a_neg, int n) {
  for (int i = blockIdx.x * blockDim.x + threadIdx.x; i < n;
       i += gridDim.x * blockDim.x)
    a_neg[i] = -__expf(a_log[i]);
}

// ---------------------------------------------------------------------------
// LayerNorm over DMODEL=512: one block (256 threads, 8 waves) per row.
// ---------------------------------------------------------------------------
__global__ void k_layernorm(const float* __restrict__ x,
                            const float* __restrict__ w,
                            const float* __restrict__ b,
                            _Float16* __restrict__ xn) {
  const int row = blockIdx.x;
  const float* px = x + (size_t)row * DMODEL;
  const int c0 = threadIdx.x, c1 = threadIdx.x + 256;
  float v0 = px[c0], v1 = px[c1];
  float s = v0 + v1, s2 = v0 * v0 + v1 * v1;
#pragma unroll
  for (int m = 1; m < 32; m <<= 1) {
    s  += __shfl_xor(s,  m, 32);
    s2 += __shfl_xor(s2, m, 32);
  }
  __shared__ float ss[8], ss2[8];
  const int wv = threadIdx.x >> 5;
  if ((threadIdx.x & 31) == 0) { ss[wv] = s; ss2[wv] = s2; }
  __syncthreads();
  s = 0.f; s2 = 0.f;
#pragma unroll
  for (int i = 0; i < 8; ++i) { s += ss[i]; s2 += ss2[i]; }
  const float mu  = s  * (1.0f / DMODEL);
  const float var = s2 * (1.0f / DMODEL) - mu * mu;
  const float inv = rsqrtf(var + 1e-5f);
  _Float16* po = xn + (size_t)row * DMODEL;
  po[c0] = (_Float16)((v0 - mu) * inv * w[c0] + b[c0]);
  po[c1] = (_Float16)((v1 - mu) * inv * w[c1] + b[c1]);
}

// ---------------------------------------------------------------------------
// WMMA GEMM, 32x32 output tile per wave (2x2 16x16 WMMA tiles).
// D[M,N] f32 = A[M,K] f16 * W[N,K]^T f16 (both row-major).
// ---------------------------------------------------------------------------
template <int K, int LDA, int LDB, int LDD>
__global__ void k_gemm_f16(const _Float16* __restrict__ A,
                           const _Float16* __restrict__ Bw,
                           float* __restrict__ D, int tilesN32) {
  const int wave = (blockIdx.x * blockDim.x + threadIdx.x) >> 5;
  const int lane = threadIdx.x & 31;
  const int low = lane & 15, hi = lane >> 4;
  const int m0 = (wave / tilesN32) << 5;
  const int n0 = (wave % tilesN32) << 5;
  float8 c00 = {}, c01 = {}, c10 = {}, c11 = {};
#pragma unroll
  for (int k0 = 0; k0 < K; k0 += 32) {
    half16 a0 = load_frag_h(A,  LDA, m0 + low,      k0, hi);
    half16 a1 = load_frag_h(A,  LDA, m0 + 16 + low, k0, hi);
    half16 b0 = load_frag_h(Bw, LDB, n0 + low,      k0, hi);
    half16 b1 = load_frag_h(Bw, LDB, n0 + 16 + low, k0, hi);
    c00 = __builtin_amdgcn_wmma_f32_16x16x32_f16(false, a0, false, b0,
                                                 (short)0, c00, false, false);
    c01 = __builtin_amdgcn_wmma_f32_16x16x32_f16(false, a0, false, b1,
                                                 (short)0, c01, false, false);
    c10 = __builtin_amdgcn_wmma_f32_16x16x32_f16(false, a1, false, b0,
                                                 (short)0, c10, false, false);
    c11 = __builtin_amdgcn_wmma_f32_16x16x32_f16(false, a1, false, b1,
                                                 (short)0, c11, false, false);
  }
#pragma unroll
  for (int r = 0; r < 8; ++r) {
    const size_t r0 = (size_t)(m0 + hi * 8 + r) * LDD;
    const size_t r1 = (size_t)(m0 + 16 + hi * 8 + r) * LDD;
    D[r0 + n0 + low]      = c00[r];
    D[r0 + n0 + 16 + low] = c01[r];
    D[r1 + n0 + low]      = c10[r];
    D[r1 + n0 + 16 + low] = c11[r];
  }
}

// ---------------------------------------------------------------------------
// dt projection: dt = softplus(x_dbl[:, :32] * W_dt^T + b_dt).
// K = 32 -> exactly one WMMA per 16x16 tile. A converted f32->f16 in regs.
// ---------------------------------------------------------------------------
__global__ void k_gemm_dt(const float* __restrict__ xdbl,      // [MROWS][64]
                          const _Float16* __restrict__ Wdt,    // [DINNER][32]
                          const float* __restrict__ b_dt,      // [DINNER]
                          float* __restrict__ dt) {            // [MROWS][DINNER]
  const int wave = (blockIdx.x * blockDim.x + threadIdx.x) >> 5;
  const int lane = threadIdx.x & 31;
  const int low = lane & 15, hi = lane >> 4;
  const int tilesN = DINNER / 16;
  const int m0 = (wave / tilesN) << 4;
  const int n0 = (wave % tilesN) << 4;

  const float* pr = xdbl + (size_t)(m0 + low) * 64 + hi * 8;
  half16 a;
#pragma unroll
  for (int i = 0; i < 8; ++i) {
    a[i]     = (_Float16)pr[i];
    a[i + 8] = (_Float16)pr[i + 16];
  }
  half16 b = load_frag_h(Wdt, DTRANK, n0 + low, 0, hi);
  float8 c = {};
  c = __builtin_amdgcn_wmma_f32_16x16x32_f16(false, a, false, b,
                                             (short)0, c, false, false);
  const int col = n0 + low;
  const float bias = b_dt[col];
#pragma unroll
  for (int r = 0; r < 8; ++r) {
    dt[(size_t)(m0 + hi * 8 + r) * DINNER + col] = softplus_fast(c[r] + bias);
  }
}

// ---------------------------------------------------------------------------
// Depthwise causal conv1d (4 taps) + bias + SiLU over xs_pre (= xz[:, :1024]).
// ---------------------------------------------------------------------------
__global__ void k_conv_silu(const float* __restrict__ xz,      // [MROWS][2048]
                            const float* __restrict__ cw,      // [DINNER][4]
                            const float* __restrict__ cb,      // [DINNER]
                            float* __restrict__ xs_f,
                            _Float16* __restrict__ xs_h) {
  const int idx = blockIdx.x * blockDim.x + threadIdx.x;       // MROWS*DINNER
  if (idx >= MROWS * DINNER) return;
  const int e = idx & (DINNER - 1);
  const int m = idx >> 10;
  const int t = m & (L_DIM - 1);
  const int mb = m - t;                                        // b * L
  float acc = cb[e];
#pragma unroll
  for (int j = 0; j < 4; ++j) {
    const int tt = t - 3 + j;
    if (tt >= 0)
      acc += xz[(size_t)(mb + tt) * (2 * DINNER) + e] * cw[e * 4 + j];
  }
  const float s = silu_fast(acc);
  xs_f[(size_t)m * DINNER + e] = s;
  xs_h[(size_t)m * DINNER + e] = (_Float16)s;
}

// ---------------------------------------------------------------------------
// Fused selective scan. Lane layout: 16 states -> 16 lanes; 2 channels/wave.
// Per step: h = exp(dt*A)*h + dt*B_t*xs ; y = sum_n h*C_t (shfl_xor tree),
// then y = (y + D*xs) * silu(z) -> f16 for the out-projection.
// ---------------------------------------------------------------------------
__global__ void k_scan(const float* __restrict__ dt,      // [MROWS][DINNER]
                       const float* __restrict__ xs,      // [MROWS][DINNER]
                       const float* __restrict__ xdbl,    // [MROWS][64]
                       const float* __restrict__ xz,      // [MROWS][2048] (z)
                       const float* __restrict__ a_neg,   // [DINNER][16]
                       const float* __restrict__ Dp,      // [DINNER]
                       _Float16* __restrict__ y_h) {      // [MROWS][DINNER]
  const int lane = threadIdx.x & 31;
  const int wave = (blockIdx.x * blockDim.x + threadIdx.x) >> 5;
  const int pair = wave * 2 + (lane >> 4);                // 0 .. B*DINNER-1
  const int n    = lane & 15;
  const int bb   = pair >> 10;                            // / DINNER
  const int e    = pair & (DINNER - 1);

  const float Aval = a_neg[e * NSTATE + n];
  const float Dval = Dp[e];
  float h = 0.0f;

  for (int t = 0; t < L_DIM; ++t) {
    const int m = bb * L_DIM + t;
    const float dtv = dt[(size_t)m * DINNER + e];
    const float xv  = xs[(size_t)m * DINNER + e];
    const float bn  = xdbl[(size_t)m * 64 + DTRANK + n];
    const float cn  = xdbl[(size_t)m * 64 + DTRANK + NSTATE + n];
    const float dA  = __expf(dtv * Aval);
    h = dA * h + dtv * bn * xv;
    float p = h * cn;
    p += __shfl_xor(p, 1, 32);
    p += __shfl_xor(p, 2, 32);
    p += __shfl_xor(p, 4, 32);
    p += __shfl_xor(p, 8, 32);
    if (n == 0) {
      const float zv = xz[(size_t)m * (2 * DINNER) + DINNER + e];
      const float yv = (p + Dval * xv) * silu_fast(zv);
      y_h[(size_t)m * DINNER + e] = (_Float16)yv;
    }
  }
}

// ---------------------------------------------------------------------------
// Out-projection + residual: d_out = y * W_out^T + x (f32), 32x32 tile/wave.
// ---------------------------------------------------------------------------
__global__ void k_gemm_out(const _Float16* __restrict__ A,    // [MROWS][DINNER]
                           const _Float16* __restrict__ Bw,   // [DMODEL][DINNER]
                           const float* __restrict__ xres,    // [MROWS][DMODEL]
                           float* __restrict__ out) {         // [MROWS][DMODEL]
  const int wave = (blockIdx.x * blockDim.x + threadIdx.x) >> 5;
  const int lane = threadIdx.x & 31;
  const int low = lane & 15, hi = lane >> 4;
  const int tilesN32 = DMODEL / 32;
  const int m0 = (wave / tilesN32) << 5;
  const int n0 = (wave % tilesN32) << 5;
  float8 c00 = {}, c01 = {}, c10 = {}, c11 = {};
#pragma unroll
  for (int k0 = 0; k0 < DINNER; k0 += 32) {
    half16 a0 = load_frag_h(A,  DINNER, m0 + low,      k0, hi);
    half16 a1 = load_frag_h(A,  DINNER, m0 + 16 + low, k0, hi);
    half16 b0 = load_frag_h(Bw, DINNER, n0 + low,      k0, hi);
    half16 b1 = load_frag_h(Bw, DINNER, n0 + 16 + low, k0, hi);
    c00 = __builtin_amdgcn_wmma_f32_16x16x32_f16(false, a0, false, b0,
                                                 (short)0, c00, false, false);
    c01 = __builtin_amdgcn_wmma_f32_16x16x32_f16(false, a0, false, b1,
                                                 (short)0, c01, false, false);
    c10 = __builtin_amdgcn_wmma_f32_16x16x32_f16(false, a1, false, b0,
                                                 (short)0, c10, false, false);
    c11 = __builtin_amdgcn_wmma_f32_16x16x32_f16(false, a1, false, b1,
                                                 (short)0, c11, false, false);
  }
#pragma unroll
  for (int r = 0; r < 8; ++r) {
    const size_t r0 = (size_t)(m0 + hi * 8 + r) * DMODEL;
    const size_t r1 = (size_t)(m0 + 16 + hi * 8 + r) * DMODEL;
    out[r0 + n0 + low]      = c00[r] + xres[r0 + n0 + low];
    out[r0 + n0 + 16 + low] = c01[r] + xres[r0 + n0 + 16 + low];
    out[r1 + n0 + low]      = c10[r] + xres[r1 + n0 + low];
    out[r1 + n0 + 16 + low] = c11[r] + xres[r1 + n0 + 16 + low];
  }
}

// ---------------------------------------------------------------------------
// Launcher
// ---------------------------------------------------------------------------
extern "C" void kernel_launch(void* const* d_in, const int* in_sizes, int n_in,
                              void* d_out, int out_size, void* d_ws, size_t ws_size,
                              hipStream_t stream) {
  const float* x      = (const float*)d_in[0];
  const float* ln_w   = (const float*)d_in[1];
  const float* ln_b   = (const float*)d_in[2];
  const float* W_in   = (const float*)d_in[3];   // [2048][512]
  const float* conv_w = (const float*)d_in[4];   // [1024][4]
  const float* conv_b = (const float*)d_in[5];
  const float* W_x    = (const float*)d_in[6];   // [64][1024]
  const float* W_dt   = (const float*)d_in[7];   // [1024][32]
  const float* b_dt   = (const float*)d_in[8];
  const float* A_log  = (const float*)d_in[9];   // [1024][16]
  const float* Dvec   = (const float*)d_in[10];
  const float* W_out  = (const float*)d_in[11];  // [512][1024]
  float* out = (float*)d_out;

  // workspace carve-up (all offsets 256B-aligned by construction)
  char* ws = (char*)d_ws;
  size_t o = 0;
  auto take = [&](size_t bytes) {
    char* p = ws + o;
    o = (o + bytes + 255) & ~(size_t)255;
    return p;
  };
  _Float16* W_in_h  = (_Float16*)take((size_t)2 * DINNER * DMODEL * 2);
  _Float16* W_x_h   = (_Float16*)take((size_t)64 * DINNER * 2);
  _Float16* W_dt_h  = (_Float16*)take((size_t)DINNER * DTRANK * 2);
  _Float16* W_out_h = (_Float16*)take((size_t)DMODEL * DINNER * 2);
  float*    a_neg   = (float*)take((size_t)DINNER * NSTATE * 4);
  _Float16* xn_h    = (_Float16*)take((size_t)MROWS * DMODEL * 2);
  float*    xz      = (float*)take((size_t)MROWS * 2 * DINNER * 4);
  float*    xs_f    = (float*)take((size_t)MROWS * DINNER * 4);
  _Float16* xs_h    = (_Float16*)take((size_t)MROWS * DINNER * 2);
  float*    xdbl    = (float*)take((size_t)MROWS * 64 * 4);
  float*    dtb     = (float*)take((size_t)MROWS * DINNER * 4);
  _Float16* y_h     = (_Float16*)take((size_t)MROWS * DINNER * 2);
  (void)ws_size; (void)in_sizes; (void)n_in; (void)out_size;

  // 1) weight conversions + A = -exp(A_log)
  k_f32_to_f16<<<1024, 256, 0, stream>>>(W_in,  W_in_h,  2 * DINNER * DMODEL);
  k_f32_to_f16<<<256,  256, 0, stream>>>(W_x,   W_x_h,   64 * DINNER);
  k_f32_to_f16<<<128,  256, 0, stream>>>(W_dt,  W_dt_h,  DINNER * DTRANK);
  k_f32_to_f16<<<512,  256, 0, stream>>>(W_out, W_out_h, DMODEL * DINNER);
  k_a_neg<<<64, 256, 0, stream>>>(A_log, a_neg, DINNER * NSTATE);

  // 2) LayerNorm -> xn (f16)
  k_layernorm<<<MROWS, 256, 0, stream>>>(x, ln_w, ln_b, xn_h);

  // 3) in-projection: xz[4096][2048] = xn * W_in^T  (M=4096,N=2048,K=512)
  {
    const int tilesN32 = (2 * DINNER) / 32;                 // 64
    const int waves = (MROWS / 32) * tilesN32;              // 8192
    k_gemm_f16<DMODEL, DMODEL, DMODEL, 2 * DINNER>
        <<<waves / 8, 256, 0, stream>>>(xn_h, W_in_h, xz, tilesN32);
  }

  // 4) causal depthwise conv + SiLU -> xs (f32 + f16)
  k_conv_silu<<<(MROWS * DINNER) / 256, 256, 0, stream>>>(xz, conv_w, conv_b,
                                                          xs_f, xs_h);

  // 5) x-projection: x_dbl[4096][64] = xs * W_x^T  (M=4096,N=64,K=1024)
  {
    const int tilesN32 = 64 / 32;                           // 2
    const int waves = (MROWS / 32) * tilesN32;              // 256
    k_gemm_f16<DINNER, DINNER, DINNER, 64>
        <<<waves / 8, 256, 0, stream>>>(xs_h, W_x_h, xdbl, tilesN32);
  }

  // 6) dt = softplus(x_dbl[:, :32] * W_dt^T + b_dt)  (M=4096,N=1024,K=32)
  {
    const int waves = (MROWS / 16) * (DINNER / 16);         // 16384
    k_gemm_dt<<<waves / 8, 256, 0, stream>>>(xdbl, W_dt_h, b_dt, dtb);
  }

  // 7) fused selective scan -> y (f16), gated by silu(z), + D*xs
  {
    const int waves = (B_DIM * DINNER) / 2;                 // 1024 waves
    k_scan<<<waves * 32 / 256, 256, 0, stream>>>(dtb, xs_f, xdbl, xz, a_neg,
                                                 Dvec, y_h);
  }

  // 8) out-projection + residual  (M=4096,N=512,K=1024)
  {
    const int waves = (MROWS / 32) * (DMODEL / 32);         // 2048
    k_gemm_out<<<waves / 8, 256, 0, stream>>>(y_h, W_out_h, x, out);
  }
}
